// Attention_35768487641569
// MI455X (gfx1250) — matrix operand order repaired and checked by reading, not compiled, not run on hardware
//
#include <hip/hip_runtime.h>
#include <hip/hip_bf16.h>
#include <cstdint>

// ---------- types ----------
typedef __bf16 bf16_t;
typedef bf16_t bx16 __attribute__((ext_vector_type(16)));
typedef bf16_t bx8  __attribute__((ext_vector_type(8)));
typedef float  fx8  __attribute__((ext_vector_type(8)));
typedef uint32_t u32x4 __attribute__((ext_vector_type(4)));
typedef int      i32x8 __attribute__((ext_vector_type(8)));
typedef int      i32x4 __attribute__((ext_vector_type(4)));

#define D_MODEL 768
#define NHEAD   12
#define DHEAD   64
#define NCTX    2048
#define BATCH   2
#define MROWS   (BATCH * NCTX)                    // 4096
#define QKV_ELEMS (BATCH * NHEAD * NCTX * DHEAD)  // 3145728

// ---------- helpers ----------
__device__ __forceinline__ uint32_t pack2bf(float a, float b) {
  union { bf16_t h[2]; uint32_t u; } v;
  v.h[0] = (bf16_t)a;
  v.h[1] = (bf16_t)b;
  return v.u;
}
__device__ __forceinline__ uint16_t bf1(float a) {
  union { bf16_t h; uint16_t u; } v;
  v.h = (bf16_t)a;
  return v.u;
}

__device__ __forceinline__ fx8 wmma_bf16(bx16 a, bx16 b, fx8 c) {
  return __builtin_amdgcn_wmma_f32_16x16x32_bf16(
      false, a, false, b, (short)0, c, false, false);
}

__device__ __forceinline__ bx16 cat8(bx8 lo, bx8 hi) {
  return __builtin_shufflevector(lo, hi, 0,1,2,3,4,5,6,7,8,9,10,11,12,13,14,15);
}

__device__ __forceinline__ float rowmax16(float v) {
  v = fmaxf(v, __shfl_xor(v, 1, 32));
  v = fmaxf(v, __shfl_xor(v, 2, 32));
  v = fmaxf(v, __shfl_xor(v, 4, 32));
  v = fmaxf(v, __shfl_xor(v, 8, 32));
  return v;
}
__device__ __forceinline__ float rowsum16(float v) {
  v += __shfl_xor(v, 1, 32);
  v += __shfl_xor(v, 2, 32);
  v += __shfl_xor(v, 4, 32);
  v += __shfl_xor(v, 8, 32);
  return v;
}

// ---------- Tensor Data Mover: 2D bf16 tile -> LDS (D# per ISA 08 §8) ----------
// tile_d1 == 0 -> 1D copy of tile_d0 elements. All dims in 2-byte elements.
// 6-arg builtin form (clang-23): (u32x4 g0, i32x8 g1, i32x4, i32x4, i32x8, cpol)
__device__ __forceinline__ void tdm_load_2d(const void* gaddr, uint32_t lds_byte,
                                            uint32_t tile_d0, uint32_t tile_d1,
                                            uint32_t tensor_d0, uint32_t tensor_d1,
                                            uint32_t stride0) {
  uint64_t ga = (uint64_t)(uintptr_t)gaddr;
  u32x4 g0;
  g0[0] = 1u;                                              // count=1 (valid), user mode
  g0[1] = lds_byte;                                        // lds_addr
  g0[2] = (uint32_t)ga;                                    // global_addr[31:0]
  g0[3] = (uint32_t)((ga >> 32) & 0x01FFFFFFu) | (2u << 30); // addr[56:32] | type=2
  i32x8 g1;
  g1[0] = (int)(1u << 16);                                 // data_size=1 -> 2 bytes
  g1[1] = (int)((tensor_d0 & 0xFFFFu) << 16);              // tensor_dim0[15:0]
  g1[2] = (int)((tensor_d0 >> 16) | ((tensor_d1 & 0xFFFFu) << 16));
  g1[3] = (int)((tensor_d1 >> 16) | (tile_d0 << 16));      // tile_dim0
  g1[4] = (int)(tile_d1 & 0xFFFFu);                        // tile_dim1 (tile_dim2=0)
  g1[5] = (int)stride0;                                    // tensor_dim0_stride[31:0]
  g1[6] = 0;
  g1[7] = 0;
  i32x4 z4 = {0, 0, 0, 0};
  i32x8 z8 = {0, 0, 0, 0, 0, 0, 0, 0};
  __builtin_amdgcn_tensor_load_to_lds(g0, g1, z4, z4, z8, 0);
}
__device__ __forceinline__ uint32_t lds_addr_of(const void* p) {
  return (uint32_t)(uintptr_t)p;   // low 32 bits of generic ptr = LDS offset
}

// ============================================================================
// Kernel 1: QKV projection (templated; WHICH==0 pre-scales W_Q by 1/sqrt(64)).
// grid (12 nTiles, 64 mTiles), block 128 (4 waves, each a 32x32 tile).
// WHICH==0/1: store [b,h,p,e]; WHICH==2: store transposed [b,h,e,p].
// ============================================================================
template <int WHICH>
__global__ __launch_bounds__(128)
void qkv_gemm_kernel(const float* __restrict__ x,
                     const float* __restrict__ W,
                     uint16_t* __restrict__ Out)
{
  __shared__ __align__(32) uint32_t lA32[64 * 16];
  __shared__ __align__(32) uint32_t lB32[64 * 16];
  uint16_t* lA = (uint16_t*)lA32;
  uint16_t* lB = (uint16_t*)lB32;

  const int mBase = blockIdx.y * 64;
  const int nBase = blockIdx.x * 64;
  const int tid  = threadIdx.x;
  const int wave = tid >> 5, lane = tid & 31;
  const int half = lane >> 4, ln = lane & 15;
  const int waveM = (wave >> 1) * 32, waveN = (wave & 1) * 32;
  const float wscale = (WHICH == 0) ? 0.125f : 1.0f;   // fold 1/sqrt(d) into Q

  fx8 acc[2][2] = {};

  for (int k0 = 0; k0 < D_MODEL; k0 += 32) {
    __syncthreads();
#pragma unroll
    for (int i = 0; i < 8; i++) {
      int pi = tid + i * 128;
      int r = pi >> 4, c = (pi & 15) * 2;
      const float* ax = &x[(size_t)(mBase + r) * D_MODEL + k0 + c];
      const float* aw = &W[(size_t)(nBase + r) * D_MODEL + k0 + c];
      lA32[pi] = pack2bf(ax[0], ax[1]);
      lB32[pi] = pack2bf(aw[0] * wscale, aw[1] * wscale);
    }
    __syncthreads();

    bx16 a[2], b[2];
#pragma unroll
    for (int t = 0; t < 2; t++) {
      int row = waveM + t * 16 + ln;
      bx8 lo = *(const bx8*)&lA[row * 32 + half * 8];
      bx8 hi = *(const bx8*)&lA[row * 32 + 16 + half * 8];
      a[t] = cat8(lo, hi);
    }
#pragma unroll
    for (int t = 0; t < 2; t++) {
      int col = waveN + t * 16 + ln;
      b[t] = cat8(*(const bx8*)&lB[col * 32 + half * 8], *(const bx8*)&lB[col * 32 + 16 + half * 8]);
    }
#pragma unroll
    for (int i = 0; i < 2; i++)
#pragma unroll
      for (int j = 0; j < 2; j++)
        acc[i][j] = wmma_bf16(a[i], b[j], acc[i][j]);
  }

#pragma unroll
  for (int i = 0; i < 2; i++)
#pragma unroll
    for (int j = 0; j < 2; j++)
#pragma unroll
      for (int r = 0; r < 8; r++) {
        int M = mBase + waveM + i * 16 + r + half * 8;
        int f = nBase + waveN + j * 16 + ln;
        int bb = M >> 11, p = M & 2047;
        int hh = f >> 6,  e = f & 63;
        uint16_t val = bf1(acc[i][j][r]);
        if (WHICH == 2)
          Out[(size_t)(((bb * NHEAD + hh) * DHEAD) + e) * NCTX + p] = val;
        else
          Out[(size_t)(((bb * NHEAD + hh) * NCTX) + p) * DHEAD + e] = val;
      }
}

// ============================================================================
// Online-softmax update; MASK instantiated only for diagonal chunks.
// ============================================================================
template <bool MASK>
__device__ __forceinline__ void softmax_update(
    const fx8 s0v, const fx8 s1v, int q0, int half, int ln, int kvB,
    float* mrow, float* lrow, fx8* zacc, float* p0v, float* p1v)
{
#pragma unroll
  for (int r = 0; r < 8; r++) {
    float s0 = s0v[r];
    float s1 = s1v[r];
    if (MASK) {
      int qrow = q0 + r + half * 8;
      if (kvB + ln > qrow)      s0 = -1e30f;
      if (kvB + 16 + ln > qrow) s1 = -1e30f;
    }
    float mx = rowmax16(fmaxf(s0, s1));
    float mnew = fmaxf(mrow[r], mx);
    float alpha = __expf(mrow[r] - mnew);
    float p0 = __expf(s0 - mnew);
    float p1 = __expf(s1 - mnew);
    lrow[r] = lrow[r] * alpha + rowsum16(p0 + p1);
    mrow[r] = mnew;
#pragma unroll
    for (int te = 0; te < 4; te++) zacc[te][r] *= alpha;
    p0v[r] = p0; p1v[r] = p1;
  }
}

// ============================================================================
// Kernel 2: causal flash attention. block 128 (4 waves); wave 0 drives the TDM
// to double-buffer 32-key K/V chunks into LDS; all waves consume via WMMA.
// ============================================================================
__global__ __launch_bounds__(128)
void attn_kernel(const uint16_t* __restrict__ Q,
                 const uint16_t* __restrict__ K,
                 const uint16_t* __restrict__ Vt,
                 uint16_t* __restrict__ Z)
{
  __shared__ __align__(32) uint16_t lK[2][32 * 64];   // [kv][e]
  __shared__ __align__(32) uint16_t lV[2][64 * 32];   // [e][kv]
  __shared__ __align__(32) uint16_t lP[4][16 * 32];   // per-wave P bounce

  const int qBase = blockIdx.x * 64;
  const int h = blockIdx.y, b = blockIdx.z;
  const int tid = threadIdx.x, wave = tid >> 5, lane = tid & 31;
  const int half = lane >> 4, ln = lane & 15;
  const int q0 = qBase + wave * 16;

  const uint16_t* Qb = Q  + (size_t)((b * NHEAD + h) * NCTX) * DHEAD;
  const uint16_t* Kb = K  + (size_t)((b * NHEAD + h) * NCTX) * DHEAD;
  const uint16_t* Vb = Vt + (size_t)((b * NHEAD + h) * DHEAD) * NCTX;

  // resident Q A-fragments (W_Q was pre-scaled by 1/sqrt(d))
  bx16 aq[2];
  {
    int row = q0 + ln;
#pragma unroll
    for (int c = 0; c < 2; c++) {
      bx8 lo = *(const bx8*)&Qb[row * DHEAD + c * 32 + half * 8];
      bx8 hi = *(const bx8*)&Qb[row * DHEAD + c * 32 + 16 + half * 8];
      aq[c] = cat8(lo, hi);
    }
  }

  fx8 zacc[4] = {};
  float mrow[8], lrow[8];
#pragma unroll
  for (int r = 0; r < 8; r++) { mrow[r] = -1e30f; lrow[r] = 0.0f; }

  const int nChunks = (qBase >> 5) + 2;

  // prologue: TDM chunk 0 into buffer 0
  if (wave == 0) {
    tdm_load_2d(Kb, lds_addr_of(&lK[0][0]), 2048, 0, 2048, 1, 2048);
    tdm_load_2d(Vb, lds_addr_of(&lV[0][0]), 32, 64, 2048, 64, 2048);
  }

  for (int ch = 0; ch < nChunks; ch++) {
    const int kvB = ch * 32;
    const int buf = ch & 1;

    if (wave == 0) __builtin_amdgcn_s_wait_tensorcnt(0);
    __syncthreads();                       // chunk ch resident; old buffer free

    if (wave == 0 && ch + 1 < nChunks) {   // pipeline chunk ch+1
      const int nB = kvB + 32;
      tdm_load_2d(Kb + (size_t)nB * DHEAD, lds_addr_of(&lK[buf ^ 1][0]),
                  2048, 0, 2048, 1, 2048);
      tdm_load_2d(Vb + nB, lds_addr_of(&lV[buf ^ 1][0]),
                  32, 64, 2048, 64, 2048);
    }

    const uint16_t* kb = lK[buf];
    const uint16_t* vb = lV[buf];

    // S = Q * K^T
    fx8 s[2] = {};
#pragma unroll
    for (int tj = 0; tj < 2; tj++)
#pragma unroll
      for (int c = 0; c < 2; c++) {
        bx16 bk = cat8(*(const bx8*)&kb[(tj * 16 + ln) * DHEAD + c * 32 + half * 8], *(const bx8*)&kb[(tj * 16 + ln) * DHEAD + c * 32 + 16 + half * 8]);
        s[tj] = wmma_bf16(aq[c], bk, s[tj]);
      }

    float p0v[8], p1v[8];
    if (kvB + 31 <= q0)
      softmax_update<false>(s[0], s[1], q0, half, ln, kvB, mrow, lrow, zacc, p0v, p1v);
    else
      softmax_update<true>(s[0], s[1], q0, half, ln, kvB, mrow, lrow, zacc, p0v, p1v);

    // P: C/D layout -> A layout via per-wave LDS
    uint16_t* lp = lP[wave];
#pragma unroll
    for (int r = 0; r < 8; r++) {
      int rr = r + half * 8;
      lp[rr * 32 + ln]      = bf1(p0v[r]);
      lp[rr * 32 + 16 + ln] = bf1(p1v[r]);
    }
    bx16 ap;
    {
      bx8 lo = *(const bx8*)&lp[ln * 32 + half * 8];
      bx8 hi = *(const bx8*)&lp[ln * 32 + 16 + half * 8];
      ap = cat8(lo, hi);
    }

    // Z += P * V
#pragma unroll
    for (int te = 0; te < 4; te++) {
      bx16 bv = cat8(*(const bx8*)&vb[(te * 16 + ln) * 32 + half * 8], *(const bx8*)&vb[(te * 16 + ln) * 32 + 16 + half * 8]);
      zacc[te] = wmma_bf16(ap, bv, zacc[te]);
    }
  }

  // normalize and store Z (bf16)
#pragma unroll
  for (int r = 0; r < 8; r++) {
    float inv = 1.0f / lrow[r];
    int qrow = q0 + r + half * 8;
#pragma unroll
    for (int te = 0; te < 4; te++) {
      int e = te * 16 + ln;
      Z[(size_t)(b * NCTX + qrow) * (NHEAD * DHEAD) + h * DHEAD + e] =
          bf1(zacc[te][r] * inv);
    }
  }
}

// ============================================================================
// Kernel 3: output projection. out[m,d] = sum_f Z[m,f] * Wo[d,f]  (f32 out)
// A tile (already bf16) staged by TDM; B tile converted manually.
// ============================================================================
__global__ __launch_bounds__(128)
void out_gemm_kernel(const uint16_t* __restrict__ Zf,
                     const float* __restrict__ Wo,
                     float* __restrict__ out)
{
  __shared__ __align__(32) uint16_t lA[64 * 32];
  __shared__ __align__(32) uint32_t lB32[64 * 16];
  uint16_t* lB = (uint16_t*)lB32;

  const int mBase = blockIdx.y * 64;
  const int nBase = blockIdx.x * 64;
  const int tid  = threadIdx.x;
  const int wave = tid >> 5, lane = tid & 31;
  const int half = lane >> 4, ln = lane & 15;
  const int waveM = (wave >> 1) * 32, waveN = (wave & 1) * 32;

  fx8 acc[2][2] = {};

  for (int k0 = 0; k0 < D_MODEL; k0 += 32) {
    __syncthreads();                       // previous tiles fully consumed
    if (wave == 0)                         // TDM: 64x32 bf16 tile of Z
      tdm_load_2d(Zf + (size_t)mBase * D_MODEL + k0, lds_addr_of(lA),
                  32, 64, D_MODEL, MROWS, D_MODEL);
#pragma unroll
    for (int i = 0; i < 8; i++) {
      int pi = tid + i * 128;
      int r = pi >> 4, c = (pi & 15) * 2;
      const float* aw = &Wo[(size_t)(nBase + r) * D_MODEL + k0 + c];
      lB32[pi] = pack2bf(aw[0], aw[1]);
    }
    if (wave == 0) __builtin_amdgcn_s_wait_tensorcnt(0);
    __syncthreads();

    bx16 a[2], b[2];
#pragma unroll
    for (int t = 0; t < 2; t++) {
      int row = waveM + t * 16 + ln;
      bx8 lo = *(const bx8*)&lA[row * 32 + half * 8];
      bx8 hi = *(const bx8*)&lA[row * 32 + 16 + half * 8];
      a[t] = cat8(lo, hi);
    }
#pragma unroll
    for (int t = 0; t < 2; t++) {
      int col = waveN + t * 16 + ln;
      b[t] = cat8(*(const bx8*)&lB[col * 32 + half * 8], *(const bx8*)&lB[col * 32 + 16 + half * 8]);
    }
#pragma unroll
    for (int i = 0; i < 2; i++)
#pragma unroll
      for (int j = 0; j < 2; j++)
        acc[i][j] = wmma_bf16(a[i], b[j], acc[i][j]);
  }

#pragma unroll
  for (int i = 0; i < 2; i++)
#pragma unroll
    for (int j = 0; j < 2; j++)
#pragma unroll
      for (int r = 0; r < 8; r++) {
        int M = mBase + waveM + i * 16 + r + half * 8;
        int f = nBase + waveN + j * 16 + ln;
        out[(size_t)M * D_MODEL + f] = acc[i][j][r];
      }
}

// ============================================================================
extern "C" void kernel_launch(void* const* d_in, const int* in_sizes, int n_in,
                              void* d_out, int out_size, void* d_ws, size_t ws_size,
                              hipStream_t stream) {
  // setup_inputs order: x, W_K, W_Q, W_V, W_O
  const float* x  = (const float*)d_in[0];
  const float* Wk = (const float*)d_in[1];
  const float* Wq = (const float*)d_in[2];
  const float* Wv = (const float*)d_in[3];
  const float* Wo = (const float*)d_in[4];
  float* out = (float*)d_out;

  uint16_t* Qbf = (uint16_t*)d_ws;               // [B,H,N,64] bf16 (pre-scaled)
  uint16_t* Kbf = Qbf + QKV_ELEMS;               // [B,H,N,64] bf16
  uint16_t* Vtb = Kbf + QKV_ELEMS;               // [B,H,64,N] bf16 (transposed)
  uint16_t* Zf  = Vtb + QKV_ELEMS;               // [B*N, 768] bf16

  dim3 ggrid(D_MODEL / 64, MROWS / 64);
  qkv_gemm_kernel<0><<<ggrid, 128, 0, stream>>>(x, Wq, Qbf);
  qkv_gemm_kernel<1><<<ggrid, 128, 0, stream>>>(x, Wk, Kbf);
  qkv_gemm_kernel<2><<<ggrid, 128, 0, stream>>>(x, Wv, Vtb);
  attn_kernel<<<dim3(NCTX / 64, NHEAD, BATCH), 128, 0, stream>>>(
      Qbf, Kbf, Vtb, Zf);
  out_gemm_kernel<<<ggrid, 128, 0, stream>>>(Zf, Wo, out);
}
